// Attention_33457795236557
// MI455X (gfx1250) — matrix operand order repaired and checked by reading, not compiled, not run on hardware
//
#include <hip/hip_runtime.h>
#include <hip/hip_bf16.h>

// Problem constants (B,H,S,D,HID) = (2,4,512,16,32)
#define B_   2
#define H_   4
#define S_   512
#define D_   16
#define HID_ 32
#define BHS_ (B_*H_*S_)

typedef __bf16 bf16;
typedef __attribute__((ext_vector_type(16))) bf16  v16bf;
typedef __attribute__((ext_vector_type(8)))  float v8f;
typedef __attribute__((ext_vector_type(4)))  unsigned int ui4;
typedef __attribute__((ext_vector_type(8)))  int i8v;
typedef __attribute__((ext_vector_type(4)))  int i4v;

__device__ __forceinline__ v8f zero8() {
    v8f z;
#pragma unroll
    for (int i = 0; i < 8; ++i) z[i] = 0.0f;
    return z;
}

// A-operand (16-bit, 16x32, M x K): lane L holds row M=L%16, half=L/16.
// element e -> K = e + 8*half + (e>=8 ? 8 : 0)   (ISA 7.12.2, 16-bit A 16x32)
__device__ __forceinline__ v16bf make_a(const float* src, int ld, int kvalid) {
    const int lane = threadIdx.x & 31;
    const int m = lane & 15, half = lane >> 4;
    v16bf a;
#pragma unroll
    for (int e = 0; e < 16; ++e) {
        const int kk = e + 8 * half + ((e >= 8) ? 8 : 0);
        const float x = (kk < kvalid) ? src[m * ld + kk] : 0.0f;
        a[e] = (bf16)x;
    }
    return a;
}

// B-operand (16-bit, 32x16, K x N): lane L holds col N=L%16, half=L/16.
// element e -> K = e + 8*half + (e>=8 ? 8 : 0)   (same k order as A)
__device__ __forceinline__ v16bf make_b(const float* src, int ld, int kvalid) {
    const int lane = threadIdx.x & 31;
    const int n = lane & 15, half = lane >> 4;
    v16bf b;
#pragma unroll
    for (int e = 0; e < 16; ++e) {
        const int kk = e + 8 * half + ((e >= 8) ? 8 : 0);
        const float x = (kk < kvalid) ? src[kk * ld + n] : 0.0f;
        b[e] = (bf16)x;
    }
    return b;
}

__device__ __forceinline__ v8f wmma_bf16(v16bf a, v16bf b, v8f c) {
    return __builtin_amdgcn_wmma_f32_16x16x32_bf16(false, a, false, b, (short)0, c,
                                                   false, false);
}

// C/D layout (f32 16x16): VGPR r -> M = r + 8*half, N = lane%16.
__device__ __forceinline__ void store_c(float* dst, int ld, v8f c) {
    const int lane = threadIdx.x & 31;
    const int n = lane & 15, half = lane >> 4;
#pragma unroll
    for (int r = 0; r < 8; ++r) dst[(r + 8 * half) * ld + n] = c[r];
}

// ---------------------------------------------------------------------------
// Tensor Data Mover descriptors (CDNA5 ISA).
// 1-D contiguous copy of `nelem` fp32 elements between global and LDS.
// Group0: [1:0]=count=1, [63:32]=lds_addr, [120:64]=global_addr, [127:126]=type=2
// Group1: data_size=4B (code 2), tensor_dim0=tile_dim0=nelem, tensor_dim1=1,
//         dim0_stride=nelem, no padding / iteration / multicast.
// All inputs are uniform (blockIdx / kernarg derived) -> SGPR operands.
// This toolchain's builtin takes 6 args: (g0, g1, g2, g3, g4, cpol).
// ---------------------------------------------------------------------------
__device__ __forceinline__ void tdm_desc(ui4& g0, i8v& g1,
                                         unsigned lds_off,
                                         unsigned long long ga,
                                         unsigned nelem) {
    g0[0] = 1u;                                         // count=1
    g0[1] = lds_off;                                    // lds_addr (bytes)
    g0[2] = (unsigned)(ga & 0xFFFFFFFFu);               // global_addr[31:0]
    g0[3] = (unsigned)((ga >> 32) & 0x01FFFFFFu) | (2u << 30);  // addr[56:32]|type=2

    g1[0] = (int)(2u << 16);                            // data_size = 4 bytes
    g1[1] = (int)((nelem & 0xFFFFu) << 16);             // tensor_dim0[15:0]
    g1[2] = (int)(((nelem >> 16) & 0xFFFFu) | (1u << 16)); // dim0[31:16] | tensor_dim1=1
    g1[3] = (int)((nelem & 0xFFFFu) << 16);             // tile_dim0 = nelem (<=8192)
    g1[4] = 0;                                          // tile_dim1/2 unused
    g1[5] = (int)nelem;                                 // tensor_dim0_stride[31:0]
    g1[6] = 0;
    g1[7] = 0;
}

__device__ __forceinline__ void tdm_load_1d(float* lds_dst, const float* gsrc,
                                            unsigned nelem) {
    ui4 g0; i8v g1;
    tdm_desc(g0, g1, (unsigned)(unsigned long long)(void*)lds_dst,
             (unsigned long long)gsrc, nelem);
    i4v z4; z4[0] = 0; z4[1] = 0; z4[2] = 0; z4[3] = 0;
    i8v z8;
#pragma unroll
    for (int i = 0; i < 8; ++i) z8[i] = 0;
    __builtin_amdgcn_tensor_load_to_lds(g0, g1, z4, z4, z8, 0);
}

__device__ __forceinline__ void tdm_store_1d(const float* lds_src, float* gdst,
                                             unsigned nelem) {
    ui4 g0; i8v g1;
    tdm_desc(g0, g1, (unsigned)(unsigned long long)(const void*)lds_src,
             (unsigned long long)gdst, nelem);
    i4v z4; z4[0] = 0; z4[1] = 0; z4[2] = 0; z4[3] = 0;
    i8v z8;
#pragma unroll
    for (int i = 0; i < 8; ++i) z8[i] = 0;
    __builtin_amdgcn_tensor_store_from_lds(g0, g1, z4, z4, z8, 0);
}

// ---------------------------------------------------------------------------
// Kernel 1: token projections + pair-MLP hidden features, one wave per
// 16-token tile. 11 WMMAs per wave. b1 folded into hqA/hqB.
// ---------------------------------------------------------------------------
__global__ __launch_bounds__(32) void proj_kernel(
    const float* __restrict__ q, const float* __restrict__ k, const float* __restrict__ v,
    const float* __restrict__ Wq, const float* __restrict__ bq,
    const float* __restrict__ Wk, const float* __restrict__ bk,
    const float* __restrict__ Wv, const float* __restrict__ bv,
    const float* __restrict__ N1, const float* __restrict__ b1,
    float* __restrict__ vp,
    float* __restrict__ hqA, float* __restrict__ hqB,
    float* __restrict__ hkA, float* __restrict__ hkB)
{
    __shared__ float sTile[16 * 16];
    const int lane = threadIdx.x & 31;
    const int bh   = blockIdx.x >> 5;   // S_/16 == 32 tiles per (b,h)
    const int tile = blockIdx.x & 31;
    const int t0   = bh * S_ + tile * 16;
    const int n    = lane & 15;

    v16bf bWq = make_b(Wq, D_, D_);
    v16bf bWk = make_b(Wk, D_, D_);
    v16bf bWv = make_b(Wv, D_, D_);

    v16bf aQ = make_a(q + (size_t)t0 * D_, D_, D_);
    v16bf aK = make_a(k + (size_t)t0 * D_, D_, D_);
    v16bf aV = make_a(v + (size_t)t0 * D_, D_, D_);

    v8f cq = wmma_bf16(aQ, bWq, zero8());
    v8f ck = wmma_bf16(aK, bWk, zero8());
    v8f cv = wmma_bf16(aV, bWv, zero8());
#pragma unroll
    for (int r = 0; r < 8; ++r) {
        cq[r] += bq[n]; ck[r] += bk[n]; cv[r] += bv[n];
    }

    store_c(vp + (size_t)t0 * D_, D_, cv);

    // Re-layout qp/kp (C layout -> A operand) through LDS
    store_c(sTile, 16, cq);
    __syncthreads();
    v16bf aQp = make_a(sTile, 16, 16);
    __syncthreads();
    store_c(sTile, 16, ck);
    __syncthreads();
    v16bf aKp = make_a(sTile, 16, 16);

    v16bf bA0 = make_b(N1,                  HID_, D_);
    v16bf bA1 = make_b(N1 + 16,             HID_, D_);
    v16bf bB0 = make_b(N1 + 16 * HID_,      HID_, D_);
    v16bf bB1 = make_b(N1 + 16 * HID_ + 16, HID_, D_);

    v8f c;
    c = wmma_bf16(aQp, bA0, zero8());
#pragma unroll
    for (int r = 0; r < 8; ++r) c[r] += b1[n];
    store_c(hqA + (size_t)t0 * HID_, HID_, c);
    c = wmma_bf16(aQp, bA1, zero8());
#pragma unroll
    for (int r = 0; r < 8; ++r) c[r] += b1[16 + n];
    store_c(hqA + (size_t)t0 * HID_ + 16, HID_, c);

    c = wmma_bf16(aQp, bB0, zero8());
#pragma unroll
    for (int r = 0; r < 8; ++r) c[r] += b1[n];
    store_c(hqB + (size_t)t0 * HID_, HID_, c);
    c = wmma_bf16(aQp, bB1, zero8());
#pragma unroll
    for (int r = 0; r < 8; ++r) c[r] += b1[16 + n];
    store_c(hqB + (size_t)t0 * HID_ + 16, HID_, c);

    c = wmma_bf16(aKp, bA0, zero8());
    store_c(hkA + (size_t)t0 * HID_, HID_, c);
    c = wmma_bf16(aKp, bA1, zero8());
    store_c(hkA + (size_t)t0 * HID_ + 16, HID_, c);
    c = wmma_bf16(aKp, bB0, zero8());
    store_c(hkB + (size_t)t0 * HID_, HID_, c);
    c = wmma_bf16(aKp, bB1, zero8());
    store_c(hkB + (size_t)t0 * HID_ + 16, HID_, c);
}

// ---------------------------------------------------------------------------
// Kernel 2: one wave per (b,h,q-tile of 16).
//   - TDM loads vp tile (32 KB) up-front, overlapped with pair-MLP compute
//   - TDM double-buffers hkA/hkB chunks (s_wait_tensorcnt pipelining)
//   - fp32 pair-MLP scores -> LDS, wave-cooperative softmax
//   - TDM stores the full 16x512 attn tile (contiguous in global)
//   - attn@vp: 16 chained v_wmma_f32_16x16x32_bf16; @Wo: 1 WMMA
// ---------------------------------------------------------------------------
__global__ __launch_bounds__(32) void attn_kernel(
    const float* __restrict__ mask,
    const float* __restrict__ N2, const float* __restrict__ b2,
    const float* __restrict__ Wo, const float* __restrict__ bo,
    const float* __restrict__ vp,
    const float* __restrict__ hqA, const float* __restrict__ hqB,
    const float* __restrict__ hkA, const float* __restrict__ hkB,
    float* __restrict__ out, float* __restrict__ attn)
{
    extern __shared__ float smem[];
    float* sLog  = smem;                   // 16*512 logits / attn probs
    float* sVp   = sLog + 16 * S_;         // 512*16 (whole vp tile)
    float* sHkA0 = sVp + S_ * D_;          // 128*32 double-buffered k chunks
    float* sHkB0 = sHkA0 + 128 * HID_;
    float* sHkA1 = sHkB0 + 128 * HID_;
    float* sHkB1 = sHkA1 + 128 * HID_;
    float* sHqA  = sHkB1 + 128 * HID_;     // 16*32
    float* sHqB  = sHqA + 16 * HID_;       // 16*32
    float* sOut  = sHqB + 16 * HID_;       // 16*16

    const int lane  = threadIdx.x & 31;
    const int bh    = blockIdx.x >> 5;
    const int qtile = blockIdx.x & 31;
    const int q0    = qtile * 16;
    const int t0    = bh * S_;
    const int b     = bh >> 2;             // H_ == 4

    // ---- async staging via Tensor Data Mover ----
    // vp tile: needed only after softmax -> fully hidden behind score compute
    tdm_load_1d(sVp, vp + (size_t)t0 * D_, S_ * D_);
    // first hk chunk into buffer 0
    tdm_load_1d(sHkA0, hkA + (size_t)t0 * HID_, 128 * HID_);
    tdm_load_1d(sHkB0, hkB + (size_t)t0 * HID_, 128 * HID_);

    // hq rows (small): plain vector loads
    for (int i = lane; i < 16 * HID_; i += 32) {
        sHqA[i] = hqA[(size_t)(t0 + q0) * HID_ + i];
        sHqB[i] = hqB[(size_t)(t0 + q0) * HID_ + i];
    }
    __syncthreads();

    const int qr   = lane & 15;            // this lane's query row
    const int part = lane >> 4;            // k-partition (even/odd)
    float hqa[HID_], hqb[HID_], n2r[HID_];
#pragma unroll
    for (int f = 0; f < HID_; ++f) {
        hqa[f] = sHqA[qr * HID_ + f];
        hqb[f] = sHqB[qr * HID_ + f];
        n2r[f] = N2[f];
    }
    const float b2v2 = 2.0f * b2[0];

    // ---- pair-MLP logits, TDM double-buffered over 4 chunks of 128 k ----
    for (int c = 0; c < 4; ++c) {
        const float* bufA = (c & 1) ? sHkA1 : sHkA0;
        const float* bufB = (c & 1) ? sHkB1 : sHkB0;
        if (c < 3) {
            float* nA = (c & 1) ? sHkA0 : sHkA1;
            float* nB = (c & 1) ? sHkB0 : sHkB1;
            tdm_load_1d(nA, hkA + (size_t)(t0 + (c + 1) * 128) * HID_, 128 * HID_);
            tdm_load_1d(nB, hkB + (size_t)(t0 + (c + 1) * 128) * HID_, 128 * HID_);
            // TDM completes in order per wave: allowing 2 outstanding (the two
            // just-issued ops) guarantees vp + chunk c are resident.
            __builtin_amdgcn_s_wait_tensorcnt(2);
        } else {
            __builtin_amdgcn_s_wait_tensorcnt(0);  // everything resident
        }
        __syncthreads();   // fence: keep LDS reads below the tensorcnt wait

        for (int kk = part; kk < 128; kk += 2) {
            const float* ha = bufA + kk * HID_;
            const float* hb = bufB + kk * HID_;
            float acc = b2v2;
#pragma unroll
            for (int f = 0; f < HID_; ++f) {
                const float p1 = hqa[f] + hb[f];   // relu(qp@A + kp@B + b1)
                const float p2 = hqb[f] + ha[f];   // relu(kp@A + qp@B + b1)^T
                acc += fmaxf(p1, 0.0f) * n2r[f];
                acc += fmaxf(p2, 0.0f) * n2r[f];
            }
            const int kg = c * 128 + kk;
            acc += mask[((size_t)b * S_ + (q0 + qr)) * S_ + kg] * -1.0e9f;
            sLog[qr * S_ + kg] = acc;
        }
        __syncthreads();
    }

    // ---- softmax over k (two lane-halves cooperate per row) ----
    float mx = -3.402823466e38f;
    for (int kk = part; kk < S_; kk += 2) mx = fmaxf(mx, sLog[qr * S_ + kk]);
    mx = fmaxf(mx, __shfl_xor(mx, 16, 32));
    float sum = 0.0f;
    for (int kk = part; kk < S_; kk += 2) {
        const float e = __expf(sLog[qr * S_ + kk] - mx);
        sLog[qr * S_ + kk] = e;
        sum += e;
    }
    sum += __shfl_xor(sum, 16, 32);
    const float inv = 1.0f / sum;
    for (int kk = part; kk < S_; kk += 2) {
        sLog[qr * S_ + kk] *= inv;
    }
    __syncthreads();

    // Entire 16x512 attn tile is contiguous in global: one TDM store,
    // overlapped with the WMMA loop below (which only *reads* sLog).
    tdm_store_1d(sLog, attn + (size_t)(t0 + q0) * S_, 16 * S_);

    // ---- attn @ vp : 16 chained v_wmma_f32_16x16x32_bf16 ----
    v8f acc8 = zero8();
#pragma unroll
    for (int kc = 0; kc < S_; kc += 32) {
        v16bf aAt = make_a(sLog + kc, S_, 32);     // 16x32 attn slice
        v16bf bV  = make_b(sVp + kc * D_, D_, 32); // 32x16 vp slice
        acc8 = wmma_bf16(aAt, bV, acc8);
    }

    // ---- output projection @ Wo + bo ----
    store_c(sOut, 16, acc8);
    __syncthreads();
    v16bf aO  = make_a(sOut, 16, 16);
    v16bf bWo = make_b(Wo, D_, D_);
    v8f co = wmma_bf16(aO, bWo, zero8());
    const int n = lane & 15, half = lane >> 4;
#pragma unroll
    for (int r = 0; r < 8; ++r)
        out[(size_t)(t0 + q0 + r + 8 * half) * D_ + n] = co[r] + bo[n];

    // Fence the async attn store (S_ENDPGM also implies wait-idle).
    __builtin_amdgcn_s_wait_tensorcnt(0);
}

// ---------------------------------------------------------------------------
extern "C" void kernel_launch(void* const* d_in, const int* in_sizes, int n_in,
                              void* d_out, int out_size, void* d_ws, size_t ws_size,
                              hipStream_t stream) {
    (void)in_sizes; (void)n_in; (void)out_size; (void)ws_size;
    const float* v    = (const float*)d_in[0];
    const float* k    = (const float*)d_in[1];
    const float* q    = (const float*)d_in[2];
    const float* mask = (const float*)d_in[3];
    const float* Wq   = (const float*)d_in[4];
    const float* bq   = (const float*)d_in[5];
    const float* Wk   = (const float*)d_in[6];
    const float* bk   = (const float*)d_in[7];
    const float* Wv   = (const float*)d_in[8];
    const float* bv   = (const float*)d_in[9];
    const float* Wo   = (const float*)d_in[10];
    const float* bo   = (const float*)d_in[11];
    const float* N1   = (const float*)d_in[12];
    const float* b1   = (const float*)d_in[13];
    const float* N2   = (const float*)d_in[14];
    const float* b2   = (const float*)d_in[15];

    float* vp  = (float*)d_ws;
    float* hqA = vp  + (size_t)BHS_ * D_;
    float* hqB = hqA + (size_t)BHS_ * HID_;
    float* hkA = hqB + (size_t)BHS_ * HID_;
    float* hkB = hkA + (size_t)BHS_ * HID_;

    float* out  = (float*)d_out;               // (B,H,S,D)
    float* attn = out + (size_t)BHS_ * D_;     // (B,H,S,S)

    const int nblocks = B_ * H_ * (S_ / 16);   // 256 waves
    proj_kernel<<<nblocks, 32, 0, stream>>>(q, k, v, Wq, bq, Wk, bk, Wv, bv,
                                            N1, b1, vp, hqA, hqB, hkA, hkB);

    // LDS: logits 32KB + vp 32KB + 4x16KB hk double buffers + hq 4KB + out 1KB
    const size_t smem = (size_t)(16 * S_ + S_ * D_ + 4 * 128 * HID_ +
                                 2 * 16 * HID_ + 16 * 16) * sizeof(float); // ~133 KB
    attn_kernel<<<nblocks, 32, smem, stream>>>(mask, N2, b2, Wo, bo, vp,
                                               hqA, hqB, hkA, hkB, out, attn);
}
